// SparseMixtureOfExperts_7894149890117
// MI455X (gfx1250) — matrix operand order repaired and checked by reading, not compiled, not run on hardware
//
#include <hip/hip_runtime.h>
#include <hip/hip_bf16.h>

// ---------------------------------------------------------------------------
// CDNA5 WMMA (wave32): D(16x16 f32) = A(16x32 bf16) x B(32x16 bf16) + C
// ---------------------------------------------------------------------------
typedef __attribute__((ext_vector_type(16))) __bf16          v16bf;
typedef __attribute__((ext_vector_type(8)))  float           v8f;
typedef __attribute__((ext_vector_type(8)))  unsigned short  v8us;

#define HH 112
#define WW 112
#define BB 16
#define NE 8
#define NF 64
#define NC 1000
#define HWPIX (HH * WW)

// Sum over the 16 lanes of each half-wave using ds_swizzle xor-mode
// (offset = xor<<10 | and 0x1F); masks < 16 never cross the half boundary.
static __device__ __forceinline__ float reduce16(float t) {
    t += __int_as_float(__builtin_amdgcn_ds_swizzle(__float_as_int(t), (1 << 10) | 0x1F));
    t += __int_as_float(__builtin_amdgcn_ds_swizzle(__float_as_int(t), (2 << 10) | 0x1F));
    t += __int_as_float(__builtin_amdgcn_ds_swizzle(__float_as_int(t), (4 << 10) | 0x1F));
    t += __int_as_float(__builtin_amdgcn_ds_swizzle(__float_as_int(t), (8 << 10) | 0x1F));
    return t;
}

// ---------------------------------------------------------------------------
// f32 -> bf16 elementwise convert
// ---------------------------------------------------------------------------
__global__ void k_f32_to_bf16(const float* __restrict__ src, __bf16* __restrict__ dst, int n) {
    int i = blockIdx.x * blockDim.x + threadIdx.x;
    if (i < n) dst[i] = (__bf16)src[i];
}

// ---------------------------------------------------------------------------
// Pre-swizzle conv weights [Cout][Cin][ks][ks] f32 into WMMA A-fragment order:
//   element (m,k): lane = m + 16*((k%16)/8), half = (k%8) + 8*(k/16)
//   dst flat index = ((mt*nkc + kc)*32 + lane)*16 + half
// ---------------------------------------------------------------------------
__global__ void k_prep_wfrag(const float* __restrict__ src, __bf16* __restrict__ dst,
                             int Cout, int Cin, int ks) {
    int kk = ks * ks;
    int Ktot = Cin * kk;
    int nkc = (Ktot + 31) >> 5;
    int mtn = (Cout + 15) >> 4;
    int total = mtn * nkc * 512;
    for (int i = blockIdx.x * blockDim.x + threadIdx.x; i < total; i += gridDim.x * blockDim.x) {
        int hh   = i & 15;
        int lane = (i >> 4) & 31;
        int kc   = (i >> 9) % nkc;
        int mt   = (i >> 9) / nkc;
        int m      = lane & 15;
        int khalf  = lane >> 4;
        int klocal = (hh & 7) + 8 * khalf + 16 * (hh >> 3);
        int kg   = kc * 32 + klocal;
        int cout = mt * 16 + m;
        float v = 0.f;
        if (kg < Ktot && cout < Cout) v = src[(size_t)(cout * Cin) * kk + kg];
        dst[i] = (__bf16)v;
    }
}

// ---------------------------------------------------------------------------
// Implicit-GEMM SAME conv, NCHW, bf16 in/out, f32 WMMA accumulate.
// Workgroup: 128 threads = 4 waves covering MT*16 Couts x 64 pixels of one
// output row. Each WAVE owns one 16-pixel N-subtile; per K-chunk it performs
// ONE B gather (2x ds_load_b128 offsets + 16 ds_load_u16 values) feeding MT
// fully-unrolled WMMAs (MT is a template constant -> static acc indexing).
// Epilogue: y = acc*scale[c] + shift[c] (+ skip) (ReLU optional).
// LDS layout: [0,512)       : float sScale[64], sShift[64]
//             [512, +2*Kpad): uint16 offtab (16B-aligned rows)
//             then          : bf16 strip[Cin*KS*66 + 66 zero slab]
// ---------------------------------------------------------------------------
template <int KS, int MT>
__global__ __launch_bounds__(128)
void k_conv_wmma(const __bf16* __restrict__ in, const __bf16* __restrict__ wfrag,
                 const float* __restrict__ scale, const float* __restrict__ shift,
                 const __bf16* __restrict__ skip, __bf16* __restrict__ out,
                 int Cin, int Cout, int relu) {
    extern __shared__ char smem[];
    constexpr int KK  = KS * KS;
    constexpr int PAD = KS >> 1;
    constexpr int CPB = MT * 16;          // channels per block

    const int Ktot = Cin * KK;
    const int nkc  = (Ktot + 31) >> 5;
    const int Kpad = nkc << 5;

    float*          sSc  = (float*)smem;
    float*          sSh  = (float*)(smem + 256);
    unsigned short* offt = (unsigned short*)(smem + 512);
    __bf16*         sIn  = (__bf16*)(smem + 512 + Kpad * 2);

    int bx   = blockIdx.x;            // b*H*2 + h*2 + span
    int span = bx & 1;
    int tmp  = bx >> 1;
    int h    = tmp % HH;
    int b    = tmp / HH;
    int w0   = span * 64;
    int c0   = blockIdx.y * CPB;      // first Cout of this block

    // ---- stage BN scale/shift for this block's channels
    if (threadIdx.x < CPB) {
        int c = c0 + threadIdx.x;
        sSc[threadIdx.x] = scale[c];
        sSh[threadIdx.x] = shift[c];
    }

    // ---- stage kg -> LDS offset table (pad entries -> zero slab)
    const int zoff = Cin * KS * 66;
    for (int kg = threadIdx.x; kg < Kpad; kg += 128) {
        int o = zoff;
        if (kg < Ktot) {
            int cin = kg / KK;
            int rem = kg - cin * KK;
            int rr  = rem / KS;
            int cc  = rem - rr * KS;
            o = (cin * KS + rr) * 66 + cc + 1 - PAD;
        }
        offt[kg] = (unsigned short)o;
    }

    // ---- stage input strip [Cin][KS rows][66 cols] + 66-entry zero slab
    int total = zoff + 66;
    for (int i = threadIdx.x; i < total; i += 128) {
        __bf16 v = (__bf16)0.f;
        if (i < zoff) {
            int s   = i % 66;
            int rr  = (i / 66) % KS;
            int cin = i / (66 * KS);
            int hin = h + rr - PAD;
            int win = w0 - 1 + s;
            if ((unsigned)hin < (unsigned)HH && (unsigned)win < (unsigned)WW)
                v = in[(((size_t)b * Cin + cin) * HH + hin) * WW + win];
        }
        sIn[i] = v;
    }
    __syncthreads();

    int wave = threadIdx.x >> 5;         // N-subtile owned by this wave
    int lane = threadIdx.x & 31;
    int n    = lane & 15;
    int khb  = (lane >> 4) * 8;
    int pb   = wave * 16 + n;            // pixel within 64-wide span

    v8f acc[MT];
#pragma unroll
    for (int mi = 0; mi < MT; ++mi) acc[mi] = (v8f){0.f,0.f,0.f,0.f,0.f,0.f,0.f,0.f};

    const __bf16* wblk = wfrag + (size_t)(blockIdx.y * MT) * nkc * 512;
    for (int kc = 0; kc < nkc; ++kc) {
        // ---- one B gather per K-chunk, reused by all MT mtiles
        const v8us* op = (const v8us*)(offt + kc * 32 + khb);
        v8us o0 = op[0], o1 = op[2];
        v16bf bf;
#pragma unroll
        for (int hh = 0; hh < 8; ++hh) bf[hh] = sIn[o0[hh] + pb];
#pragma unroll
        for (int hh = 0; hh < 8; ++hh) bf[8 + hh] = sIn[o1[hh] + pb];

        if (kc + 1 < nkc)
            __builtin_prefetch(wblk + (size_t)(kc + 1) * 512 + lane * 16, 0, 0);

#pragma unroll
        for (int mi = 0; mi < MT; ++mi) {
            v16bf a = *(const v16bf*)(wblk + ((size_t)mi * nkc + kc) * 512 + lane * 16);
            acc[mi] = __builtin_amdgcn_wmma_f32_16x16x32_bf16(false, a, false, bf,
                                                              (short)0, acc[mi], false, false);
        }
    }

    // ---- epilogue: BN + optional skip + optional ReLU, bf16 store
    int Mb = (lane >> 4) * 8;
    int w  = w0 + pb;
    if (w < WW) {
#pragma unroll
        for (int mi = 0; mi < MT; ++mi) {
#pragma unroll
            for (int j = 0; j < 8; ++j) {
                int lc   = mi * 16 + Mb + j;   // 0..CPB-1 within block
                int cout = c0 + lc;
                float v = acc[mi][j] * sSc[lc] + sSh[lc];
                size_t oidx = (((size_t)b * Cout + cout) * HH + h) * WW + w;
                if (skip) v += (float)skip[oidx];
                if (relu) v = v > 0.f ? v : 0.f;
                out[oidx] = (__bf16)v;
            }
        }
    }
}

// ---------------------------------------------------------------------------
// Expert conv (Cin=3, Cout=E*F=512): conv + bias + ReLU, fused global pooling.
// Same wave-per-N-subtile tiling; ReLU'd pixel values are reduced with
// ds_swizzle, combined across waves in LDS, one atomicAdd per channel into
// ep[B][E*F].
// LDS layout: [0,256): float sBias[64]; [256,1280): float sRed[4][64];
//             [1280,+64): uint16 offtab[32]; then bf16 strip.
// ---------------------------------------------------------------------------
__global__ __launch_bounds__(128)
void k_conv_expert_pool(const __bf16* __restrict__ in, const __bf16* __restrict__ wfrag,
                        const float* __restrict__ ebias, float* __restrict__ ep) {
    extern __shared__ char smem[];
    constexpr int KS = 3, KK = 9, PAD = 1;
    constexpr int Cin = 3;
    constexpr int Ktot = Cin * KK;          // 27
    constexpr int Kpad = 32;

    float*          sB   = (float*)smem;
    float*          sRed = (float*)(smem + 256);
    unsigned short* offt = (unsigned short*)(smem + 1280);
    __bf16*         sIn  = (__bf16*)(smem + 1280 + Kpad * 2);

    int bx   = blockIdx.x;
    int span = bx & 1;
    int tmp  = bx >> 1;
    int h    = tmp % HH;
    int b    = tmp / HH;
    int w0   = span * 64;
    int c0   = blockIdx.y * 64;

    if (threadIdx.x < 64) sB[threadIdx.x] = ebias[c0 + threadIdx.x];

    const int zoff = Cin * KS * 66;         // 594
    for (int kg = threadIdx.x; kg < Kpad; kg += 128) {
        int o = zoff;
        if (kg < Ktot) {
            int cin = kg / KK;
            int rem = kg - cin * KK;
            int rr  = rem / KS;
            int cc  = rem - rr * KS;
            o = (cin * KS + rr) * 66 + cc;
        }
        offt[kg] = (unsigned short)o;
    }

    int total = zoff + 66;
    for (int i = threadIdx.x; i < total; i += 128) {
        __bf16 v = (__bf16)0.f;
        if (i < zoff) {
            int s   = i % 66;
            int rr  = (i / 66) % KS;
            int cin = i / (66 * KS);
            int hin = h + rr - PAD;
            int win = w0 - 1 + s;
            if ((unsigned)hin < (unsigned)HH && (unsigned)win < (unsigned)WW)
                v = in[(((size_t)b * Cin + cin) * HH + hin) * WW + win];
        }
        sIn[i] = v;
    }
    __syncthreads();

    int wave = threadIdx.x >> 5;
    int lane = threadIdx.x & 31;
    int n    = lane & 15;
    int khb  = (lane >> 4) * 8;
    int pb   = wave * 16 + n;

    v8f acc[4];
#pragma unroll
    for (int mi = 0; mi < 4; ++mi) acc[mi] = (v8f){0.f,0.f,0.f,0.f,0.f,0.f,0.f,0.f};

    const v8us* op = (const v8us*)(offt + khb);
    v8us o0 = op[0], o1 = op[2];
    v16bf bf;
#pragma unroll
    for (int hh = 0; hh < 8; ++hh) bf[hh] = sIn[o0[hh] + pb];
#pragma unroll
    for (int hh = 0; hh < 8; ++hh) bf[8 + hh] = sIn[o1[hh] + pb];

    const __bf16* wblk = wfrag + (size_t)(blockIdx.y * 4) * 512;
#pragma unroll
    for (int mi = 0; mi < 4; ++mi) {
        v16bf a = *(const v16bf*)(wblk + (size_t)mi * 512 + lane * 16);
        acc[mi] = __builtin_amdgcn_wmma_f32_16x16x32_bf16(false, a, false, bf,
                                                          (short)0, acc[mi], false, false);
    }

    // ---- bias + ReLU, reduce 16 pixels per cout within wave (ds_swizzle),
    // then across waves in LDS, one atomicAdd per channel.
    int Mb = (lane >> 4) * 8;
    int w  = w0 + pb;
#pragma unroll
    for (int mi = 0; mi < 4; ++mi) {
#pragma unroll
        for (int j = 0; j < 8; ++j) {
            int lc = mi * 16 + Mb + j;
            float t = acc[mi][j] + sB[lc];
            t = t > 0.f ? t : 0.f;
            if (w >= WW) t = 0.f;
            t = reduce16(t);
            if ((lane & 15) == 0) sRed[wave * 64 + lc] = t;
        }
    }
    __syncthreads();
    if (threadIdx.x < 64) {
        int lc = threadIdx.x;
        float t = sRed[lc] + sRed[64 + lc] + sRed[128 + lc] + sRed[192 + lc];
        atomicAdd(&ep[(size_t)b * 512 + c0 + lc], t);
    }
}

// ---------------------------------------------------------------------------
// Global average pool of final gating feature map: [16,128,112,112] -> [16,128]
// ---------------------------------------------------------------------------
__global__ __launch_bounds__(256)
void k_pool(const __bf16* __restrict__ in, float* __restrict__ pooled) {
    __shared__ float red[256];
    int bc = blockIdx.x;
    const __bf16* src = in + (size_t)bc * HWPIX;
    float s = 0.f;
    for (int i = threadIdx.x; i < HWPIX; i += 256) s += (float)src[i];
    red[threadIdx.x] = s;
    __syncthreads();
    for (int off = 128; off > 0; off >>= 1) {
        if (threadIdx.x < off) red[threadIdx.x] += red[threadIdx.x + off];
        __syncthreads();
    }
    if (threadIdx.x == 0) pooled[bc] = red[0] * (1.f / (float)HWPIX);
}

// ---------------------------------------------------------------------------
// Gate: FC [16,128]->[16,8], softmax, top-2 mask, renormalize.
// ---------------------------------------------------------------------------
__global__ void k_gate(const float* __restrict__ pooled, const float* __restrict__ gfw,
                       const float* __restrict__ gfb, float* __restrict__ gwts) {
    int t = threadIdx.x;
    if (t >= BB) return;
    float lg[NE];
    for (int e = 0; e < NE; ++e) {
        float s = gfb[e];
        for (int k = 0; k < 128; ++k) s += pooled[t * 128 + k] * gfw[e * 128 + k];
        lg[e] = s;
    }
    float mx = lg[0];
    for (int e = 1; e < NE; ++e) mx = lg[e] > mx ? lg[e] : mx;
    float sum = 0.f;
    for (int e = 0; e < NE; ++e) { lg[e] = __expf(lg[e] - mx); sum += lg[e]; }
    for (int e = 0; e < NE; ++e) lg[e] /= sum;
    int i1 = 0;
    for (int e = 1; e < NE; ++e) if (lg[e] > lg[i1]) i1 = e;
    int i2 = (i1 == 0) ? 1 : 0;
    for (int e = 0; e < NE; ++e) if (e != i1 && lg[e] > lg[i2]) i2 = e;
    float tot = lg[i1] + lg[i2];
    for (int e = 0; e < NE; ++e)
        gwts[t * NE + e] = (e == i1 || e == i2) ? lg[e] / tot : 0.f;
}

// ---------------------------------------------------------------------------
// Expert head via WMMA: acc[b][c] = sum_e gwts[b][e]*(ep[b][e][:]/HW . efw[e][c][:] + efb[e][c])
// Gate weight & 1/HW folded into the B fragment; K=F=64 -> 2 bf16 chunks.
// ---------------------------------------------------------------------------
__global__ __launch_bounds__(32)
void k_head_wmma(const float* __restrict__ ep, const float* __restrict__ gwts,
                 const float* __restrict__ efw, const float* __restrict__ efb,
                 float* __restrict__ accout) {
    int ct    = blockIdx.x;
    int lane  = threadIdx.x;
    int m     = lane & 15;
    int khalf = lane >> 4;
    int bidx  = lane & 15;
    const float invHW = 1.f / (float)HWPIX;

    v8f acc = {0.f,0.f,0.f,0.f,0.f,0.f,0.f,0.f};
    for (int e = 0; e < NE; ++e) {
        float g = gwts[bidx * NE + e] * invHW;
        for (int kc = 0; kc < 2; ++kc) {
            v16bf a, bf;
#pragma unroll
            for (int hh = 0; hh < 16; ++hh) {
                int klocal = (hh & 7) + 8 * khalf + 16 * (hh >> 3);
                int fA = kc * 32 + klocal;
                int c  = ct * 16 + m;
                a[hh] = (c < NC) ? (__bf16)efw[((size_t)e * NC + c) * NF + fA] : (__bf16)0.f;
                int fB = kc * 32 + klocal;
                bf[hh] = (__bf16)(ep[((size_t)bidx * NE + e) * NF + fB] * g);
            }
            acc = __builtin_amdgcn_wmma_f32_16x16x32_bf16(false, a, false, bf,
                                                          (short)0, acc, false, false);
        }
    }
    int Mb = khalf * 8;
#pragma unroll
    for (int j = 0; j < 8; ++j) {
        int c = ct * 16 + Mb + j;
        if (c < NC) {
            float bias = 0.f;
            for (int e = 0; e < NE; ++e) bias += gwts[bidx * NE + e] * efb[e * NC + c];
            accout[bidx * NC + c] = acc[j] + bias;
        }
    }
}

// ---------------------------------------------------------------------------
// Row softmax over 1000 classes.
// ---------------------------------------------------------------------------
__global__ __launch_bounds__(256)
void k_softmax_out(const float* __restrict__ accin, float* __restrict__ out) {
    __shared__ float red[256];
    int b = blockIdx.x;
    const float* row = accin + (size_t)b * NC;
    float mx = -1e30f;
    for (int i = threadIdx.x; i < NC; i += 256) mx = row[i] > mx ? row[i] : mx;
    red[threadIdx.x] = mx;
    __syncthreads();
    for (int off = 128; off > 0; off >>= 1) {
        if (threadIdx.x < off)
            red[threadIdx.x] = red[threadIdx.x] > red[threadIdx.x + off] ? red[threadIdx.x] : red[threadIdx.x + off];
        __syncthreads();
    }
    mx = red[0];
    __syncthreads();
    float s = 0.f;
    for (int i = threadIdx.x; i < NC; i += 256) s += __expf(row[i] - mx);
    red[threadIdx.x] = s;
    __syncthreads();
    for (int off = 128; off > 0; off >>= 1) {
        if (threadIdx.x < off) red[threadIdx.x] += red[threadIdx.x + off];
        __syncthreads();
    }
    float inv = 1.f / red[0];
    for (int i = threadIdx.x; i < NC; i += 256)
        out[(size_t)b * NC + i] = __expf(row[i] - mx) * inv;
}

// ---------------------------------------------------------------------------
// Host launcher
// ---------------------------------------------------------------------------
extern "C" void kernel_launch(void* const* d_in, const int* in_sizes, int n_in,
                              void* d_out, int out_size, void* d_ws, size_t ws_size,
                              hipStream_t stream) {
    const float* x    = (const float*)d_in[0];
    const float* gw0  = (const float*)d_in[1];
    const float* gs0  = (const float*)d_in[2];
    const float* gt0  = (const float*)d_in[3];
    const float* r1w1 = (const float*)d_in[4];
    const float* r1s1 = (const float*)d_in[5];
    const float* r1t1 = (const float*)d_in[6];
    const float* r1w2 = (const float*)d_in[7];
    const float* r1s2 = (const float*)d_in[8];
    const float* r1t2 = (const float*)d_in[9];
    const float* r2w1 = (const float*)d_in[10];
    const float* r2s1 = (const float*)d_in[11];
    const float* r2t1 = (const float*)d_in[12];
    const float* r2w2 = (const float*)d_in[13];
    const float* r2s2 = (const float*)d_in[14];
    const float* r2t2 = (const float*)d_in[15];
    const float* r2ws = (const float*)d_in[16];
    const float* r2ss = (const float*)d_in[17];
    const float* r2ts = (const float*)d_in[18];
    const float* r3w1 = (const float*)d_in[19];
    const float* r3s1 = (const float*)d_in[20];
    const float* r3t1 = (const float*)d_in[21];
    const float* r3w2 = (const float*)d_in[22];
    const float* r3s2 = (const float*)d_in[23];
    const float* r3t2 = (const float*)d_in[24];
    const float* r3ws = (const float*)d_in[25];
    const float* r3ss = (const float*)d_in[26];
    const float* r3ts = (const float*)d_in[27];
    const float* gfw  = (const float*)d_in[28];
    const float* gfb  = (const float*)d_in[29];
    const float* ew   = (const float*)d_in[30];
    const float* eb   = (const float*)d_in[31];
    const float* efw  = (const float*)d_in[32];
    const float* efb  = (const float*)d_in[33];
    float* outp = (float*)d_out;

    // ---- workspace carve-up
    size_t off = 0;
    auto carve = [&](size_t bytes) { size_t o = off; off = (off + bytes + 255) & ~(size_t)255; return o; };
    char* ws = (char*)d_ws;

    const size_t nXB = (size_t)BB * 3 * HWPIX;
    const size_t nR  = (size_t)BB * 128 * HWPIX;

    __bf16* xb = (__bf16*)(ws + carve(nXB * 2));
    __bf16* R0 = (__bf16*)(ws + carve(nR * 2));
    __bf16* R1 = (__bf16*)(ws + carve(nR * 2));
    __bf16* R2 = (__bf16*)(ws + carve(nR * 2));
    __bf16* R3 = (__bf16*)(ws + carve(nR * 2));

    __bf16* wfGW0  = (__bf16*)(ws + carve(1024   * 2));
    __bf16* wfR1W1 = (__bf16*)(ws + carve(9216   * 2));
    __bf16* wfR1W2 = (__bf16*)(ws + carve(9216   * 2));
    __bf16* wfR2W1 = (__bf16*)(ws + carve(18432  * 2));
    __bf16* wfR2W2 = (__bf16*)(ws + carve(36864  * 2));
    __bf16* wfR2WS = (__bf16*)(ws + carve(2048   * 2));
    __bf16* wfR3W1 = (__bf16*)(ws + carve(73728  * 2));
    __bf16* wfR3W2 = (__bf16*)(ws + carve(147456 * 2));
    __bf16* wfR3WS = (__bf16*)(ws + carve(8192   * 2));
    __bf16* wfEW   = (__bf16*)(ws + carve(16384  * 2));

    float* ep     = (float*)(ws + carve((size_t)BB * 512 * 4));
    float* pooled = (float*)(ws + carve((size_t)BB * 128 * 4));
    float* gwts   = (float*)(ws + carve((size_t)BB * NE * 4));
    float* accbuf = (float*)(ws + carve((size_t)BB * NC * 4));

    // ---- weight prep (bf16 A-fragment swizzle)
    auto prep = [&](const float* src, __bf16* dst, int Cout, int Cin, int ks) {
        int nkc = (Cin * ks * ks + 31) / 32;
        int total = ((Cout + 15) / 16) * nkc * 512;
        hipLaunchKernelGGL(k_prep_wfrag, dim3((total + 255) / 256), dim3(256), 0, stream,
                           src, dst, Cout, Cin, ks);
    };
    prep(gw0,  wfGW0,  32, 3,   3);
    prep(r1w1, wfR1W1, 32, 32,  3);
    prep(r1w2, wfR1W2, 32, 32,  3);
    prep(r2w1, wfR2W1, 64, 32,  3);
    prep(r2w2, wfR2W2, 64, 64,  3);
    prep(r2ws, wfR2WS, 64, 32,  1);
    prep(r3w1, wfR3W1, 128, 64, 3);
    prep(r3w2, wfR3W2, 128, 128,3);
    prep(r3ws, wfR3WS, 128, 64, 1);
    prep(ew,   wfEW,   512, 3,  3);

    // ---- input to bf16
    hipLaunchKernelGGL(k_f32_to_bf16, dim3((int)((nXB + 255) / 256)), dim3(256), 0, stream,
                       x, xb, (int)nXB);

    // ---- gating CNN (WMMA implicit-GEMM convs)
    const int gx = BB * HH * 2;  // 64-pixel spans
    auto conv = [&](const __bf16* in, const __bf16* wf, const float* sc, const float* sh,
                    const __bf16* skip, __bf16* out, int Cin, int Cout, int ks, int relu) {
        int nkc = (Cin * ks * ks + 31) / 32;
        size_t lds = 512 + (size_t)nkc * 32 * 2 + ((size_t)Cin * ks * 66 + 66) * 2;
        if (ks == 3) {
            if (Cout == 32)
                hipLaunchKernelGGL((k_conv_wmma<3, 2>), dim3(gx, 1), dim3(128), lds, stream,
                                   in, wf, sc, sh, skip, out, Cin, Cout, relu);
            else
                hipLaunchKernelGGL((k_conv_wmma<3, 4>), dim3(gx, Cout / 64), dim3(128), lds, stream,
                                   in, wf, sc, sh, skip, out, Cin, Cout, relu);
        } else {
            hipLaunchKernelGGL((k_conv_wmma<1, 4>), dim3(gx, Cout / 64), dim3(128), lds, stream,
                               in, wf, sc, sh, skip, out, Cin, Cout, relu);
        }
    };

    conv(xb, wfGW0,  gs0, gt0, nullptr, R0, 3,  32, 3, 1);       // conv0 + bn + relu
    conv(R0, wfR1W1, r1s1, r1t1, nullptr, R1, 32, 32, 3, 1);     // r1 conv1
    conv(R1, wfR1W2, r1s2, r1t2, R0,      R2, 32, 32, 3, 1);     // r1 conv2 + skip + relu
    conv(R2, wfR2W1, r2s1, r2t1, nullptr, R0, 32, 64, 3, 1);     // r2 conv1
    conv(R2, wfR2WS, r2ss, r2ts, nullptr, R1, 32, 64, 1, 0);     // r2 1x1 skip
    conv(R0, wfR2W2, r2s2, r2t2, R1,      R3, 64, 64, 3, 1);     // r2 conv2 + skip + relu
    conv(R3, wfR3W1, r3s1, r3t1, nullptr, R0, 64, 128, 3, 1);    // r3 conv1
    conv(R3, wfR3WS, r3ss, r3ts, nullptr, R1, 64, 128, 1, 0);    // r3 1x1 skip
    conv(R0, wfR3W2, r3s2, r3t2, R1,      R2, 128, 128, 3, 1);   // r3 conv2 + skip + relu

    // ---- gating pool + gate weights
    hipLaunchKernelGGL(k_pool, dim3(BB * 128), dim3(256), 0, stream, R2, pooled);
    hipLaunchKernelGGL(k_gate, dim3(1), dim3(32), 0, stream, pooled, gfw, gfb, gwts);

    // ---- dense expert conv with fused global pooling
    hipMemsetAsync(ep, 0, (size_t)BB * 512 * sizeof(float), stream);
    {
        size_t lds = 1280 + 64 + ((size_t)3 * 3 * 66 + 66) * 2;
        hipLaunchKernelGGL(k_conv_expert_pool, dim3(gx, 8), dim3(128), lds, stream,
                           xb, wfEW, eb, ep);
    }

    // ---- expert head (WMMA) + final softmax
    hipLaunchKernelGGL(k_head_wmma, dim3((NC + 15) / 16), dim3(32), 0, stream,
                       ep, gwts, efw, efb, accbuf);
    hipLaunchKernelGGL(k_softmax_out, dim3(BB), dim3(256), 0, stream, accbuf, outp);

    (void)in_sizes; (void)n_in; (void)out_size; (void)ws_size;
}
